// SequenceModel_54211077210712
// MI455X (gfx1250) — compile-verified, matrix operand order checked by reading, not run on hardware
//
#include <hip/hip_runtime.h>
#include <math.h>

// ---------------- model constants ----------------
#define S_CTX   2048
#define S_TOT   2050
#define SKV     2080          // keys/rows padded to multiple of 32
#define NB      8
#define NH      4
#define DH      128
#define DMODEL  512
#define DFF     2048
#define ROWS    (NB * S_TOT)  // 16400
#define NUM_CAT 50000
#define D_PART  256
#define D_CONT  128
#define NUM_CONT 4096

typedef __attribute__((ext_vector_type(16))) __bf16 v16bf;
typedef __attribute__((ext_vector_type(8)))  __bf16 v8bf;
typedef __attribute__((ext_vector_type(8)))  float  v8f;

// ---------------- helpers ----------------
__device__ __forceinline__ __bf16 f2bf(float f) {
  union { float f; unsigned u; } v; v.f = f;
  unsigned r = v.u + 0x7fffu + ((v.u >> 16) & 1u);   // RNE
  unsigned short s = (unsigned short)(r >> 16);
  __bf16 o; __builtin_memcpy(&o, &s, 2);
  return o;
}

__device__ __forceinline__ v8f vzero8() {
  v8f v;
#pragma unroll
  for (int i = 0; i < 8; i++) v[i] = 0.f;
  return v;
}

// CDNA5 16-bit fragment layout: lane half 0 holds K {0..7,16..23},
// half 1 holds K {8..15,24..31} of a 16x32 (A) / 32x16 (B) tile.
__device__ __forceinline__ v16bf load_frag(const __bf16* rowptr, int half) {
  v8bf lo = *(const v8bf*)(rowptr + half * 8);
  v8bf hi = *(const v8bf*)(rowptr + 16 + half * 8);
  v16bf r;
#pragma unroll
  for (int i = 0; i < 8; i++) { r[i] = lo[i]; r[8 + i] = hi[i]; }
  return r;
}

__device__ __forceinline__ v8f wmma_bf16(v16bf a, v16bf b, v8f c) {
  return __builtin_amdgcn_wmma_f32_16x16x32_bf16(false, a, false, b,
                                                 (short)0, c, false, false);
}

// ---------------- elementwise kernels ----------------
__global__ void conv_bf16(const float* __restrict__ in, __bf16* __restrict__ out, int n) {
  int i = blockIdx.x * blockDim.x + threadIdx.x;
  if (i < n) out[i] = f2bf(in[i]);
}

__global__ void gen_wcont(__bf16* __restrict__ WO) {
  int idx = blockIdx.x * blockDim.x + threadIdx.x;
  if (idx >= NUM_CONT * D_CONT) return;
  int n = idx / D_CONT, i = idx % D_CONT;
  float ang = (float)n * __powf(1000.f, -(float)i / (float)D_CONT);
  WO[idx] = f2bf((i & 1) ? cosf(ang) : sinf(ang));
}

__global__ void build_x(const int* __restrict__ cat, const int* __restrict__ tim,
                        const int* __restrict__ f0, const float* __restrict__ wcat,
                        const float* __restrict__ pad,
                        float* __restrict__ XF, __bf16* __restrict__ XB) {
  int row = blockIdx.x;                       // 0..ROWS-1
  int b = row / S_TOT, st = row % S_TOT;
  size_t base = (size_t)row * DMODEL;
  for (int col = threadIdx.x; col < DMODEL; col += blockDim.x) {
    float v;
    if (st >= S_CTX) {
      v = pad[col];
    } else if (col < D_PART) {
      int c = cat[b * S_CTX + st];
      v = wcat[(size_t)c * D_PART + col];
    } else {
      bool isT = (col < D_PART + D_CONT);
      int i = isT ? (col - D_PART) : (col - D_PART - D_CONT);
      int vi = isT ? tim[b * S_CTX + st] : f0[b * S_CTX + st];
      float ang = (float)vi * __powf(10000.f, -(float)i / (float)D_CONT);
      v = (i & 1) ? cosf(ang) : sinf(ang);
    }
    XF[base + col] = v;
    XB[base + col] = f2bf(v);
  }
}

__global__ void head_gather(const float* __restrict__ XF, __bf16* __restrict__ HIN) {
  int idx = blockIdx.x * blockDim.x + threadIdx.x;   // 16*512
  if (idx >= 16 * DMODEL) return;
  int row = idx / DMODEL, col = idx % DMODEL;
  int b = row >> 1, p = row & 1;
  HIN[idx] = f2bf(XF[((size_t)(b * S_TOT + S_CTX + p)) * DMODEL + col]);
}

// x = LN(x + delta); writes fp32 and bf16 copies. One block per row.
__global__ void residual_ln(const float* __restrict__ xin, const float* __restrict__ delta,
                            const float* __restrict__ g, const float* __restrict__ bb,
                            float* __restrict__ xout, __bf16* __restrict__ xbf) {
  __shared__ float red[256];
  int row = blockIdx.x, t = threadIdx.x;
  size_t base = (size_t)row * DMODEL;
  float a0 = xin[base + t] + delta[base + t];
  float a1 = xin[base + 256 + t] + delta[base + 256 + t];
  red[t] = a0 + a1; __syncthreads();
  for (int s = 128; s > 0; s >>= 1) { if (t < s) red[t] += red[t + s]; __syncthreads(); }
  float mean = red[0] * (1.0f / DMODEL);
  __syncthreads();
  float d0 = a0 - mean, d1 = a1 - mean;
  red[t] = d0 * d0 + d1 * d1; __syncthreads();
  for (int s = 128; s > 0; s >>= 1) { if (t < s) red[t] += red[t + s]; __syncthreads(); }
  float rstd = rsqrtf(red[0] * (1.0f / DMODEL) + 1e-5f);
  float o0 = d0 * rstd * g[t] + bb[t];
  float o1 = d1 * rstd * g[256 + t] + bb[256 + t];
  xout[base + t] = o0;       xout[base + 256 + t] = o1;
  xbf[base + t] = f2bf(o0);  xbf[base + 256 + t] = f2bf(o1);
}

// ---------------- QKV GEMM with scatter epilogue ----------------
// qkv = XB(16400x512) @ W(1536x512)^T + bias; scatter q/k -> [b,h,s,d], v -> [b,h,d,s].
__global__ void __launch_bounds__(128) gemm_qkv(
    const __bf16* __restrict__ A, const __bf16* __restrict__ W,
    const float* __restrict__ bias,
    __bf16* __restrict__ Qb, __bf16* __restrict__ Kb, __bf16* __restrict__ Vt) {
  int wid  = (blockIdx.x * blockDim.x + threadIdx.x) >> 5;
  int lane = threadIdx.x & 31;
  int half = lane >> 4, lr = lane & 15;
  const int NT64 = (3 * DMODEL) / 64;   // 24
  int mt = wid / NT64, nt = wid % NT64;
  if (mt >= ROWS / 16) return;
  int m0 = mt * 16, n0 = nt * 64;
  const __bf16* Arow = A + (size_t)(m0 + lr) * DMODEL;
  const __bf16* Brow[4];
#pragma unroll
  for (int s = 0; s < 4; s++) Brow[s] = W + (size_t)(n0 + s * 16 + lr) * DMODEL;

  v8f acc[4];
#pragma unroll
  for (int s = 0; s < 4; s++) acc[s] = vzero8();

  // ping-pong double buffered K loop (NK is even: no rotation copies)
  const int NK = DMODEL / 32;   // 16
  v16bf aA = load_frag(Arow, half);
  v16bf bA[4], aB, bB[4];
#pragma unroll
  for (int s = 0; s < 4; s++) bA[s] = load_frag(Brow[s], half);
  for (int it = 0; it < NK; it += 2) {
    {
      int kc = (it + 1) * 32;
      aB = load_frag(Arow + kc, half);
#pragma unroll
      for (int s = 0; s < 4; s++) bB[s] = load_frag(Brow[s] + kc, half);
    }
#pragma unroll
    for (int s = 0; s < 4; s++) acc[s] = wmma_bf16(aA, bA[s], acc[s]);
    if (it + 2 < NK) {
      int kc = (it + 2) * 32;
      aA = load_frag(Arow + kc, half);
#pragma unroll
      for (int s = 0; s < 4; s++) bA[s] = load_frag(Brow[s] + kc, half);
    }
#pragma unroll
    for (int s = 0; s < 4; s++) acc[s] = wmma_bf16(aB, bB[s], acc[s]);
  }

#pragma unroll
  for (int s = 0; s < 4; s++) {
    int n = n0 + s * 16 + lr;
    float bv = bias[n];
#pragma unroll
    for (int r = 0; r < 8; r++) {
      int m = m0 + r + 8 * half;
      int b = m / S_TOT, st = m % S_TOT;
      __bf16 v = f2bf(acc[s][r] + bv);
      if (n < DMODEL) {
        int hh = n >> 7, d = n & 127;
        Qb[(((size_t)b * NH + hh) * SKV + st) * DH + d] = v;
      } else if (n < 2 * DMODEL) {
        int n2 = n - DMODEL; int hh = n2 >> 7, d = n2 & 127;
        Kb[(((size_t)b * NH + hh) * SKV + st) * DH + d] = v;
      } else {
        int n2 = n - 2 * DMODEL; int hh = n2 >> 7, d = n2 & 127;
        Vt[(((size_t)b * NH + hh) * DH + d) * SKV + st] = v;   // transposed
      }
    }
  }
}

// ---------------- flash attention ----------------
// one wave per (b,h,16-query tile); online softmax; P staged via LDS.
__global__ void __launch_bounds__(128) flash_attn(
    const __bf16* __restrict__ Qb, const __bf16* __restrict__ Kb,
    const __bf16* __restrict__ Vt, __bf16* __restrict__ AO) {
  __shared__ __attribute__((aligned(16))) __bf16 Pld[4][16][32];
  int widx = threadIdx.x >> 5;
  int lane = threadIdx.x & 31;
  int half = lane >> 4, lr = lane & 15;
  int wid = blockIdx.x * 4 + widx;
  const int QT = (S_TOT + 15) / 16;      // 129
  int bh = wid / QT, qt = wid % QT;
  if (bh >= NB * NH) return;
  int b = bh >> 2, h = bh & 3;
  int m0 = qt * 16;
  const float scale = 0.0883883476483184f;   // 1/sqrt(128)

  v16bf qf[4];
  const __bf16* qbase = Qb + ((size_t)bh * SKV + (m0 + lr)) * DH;
#pragma unroll
  for (int c = 0; c < 4; c++) qf[c] = load_frag(qbase + c * 32, half);

  v8f oacc[8];
#pragma unroll
  for (int c = 0; c < 8; c++) oacc[c] = vzero8();
  float mrun[8], lrun[8];
#pragma unroll
  for (int r = 0; r < 8; r++) { mrun[r] = -3.0e38f; lrun[r] = 0.f; }

  int ktmax = (m0 + 15) >> 5;
  if (ktmax > SKV / 32 - 1) ktmax = SKV / 32 - 1;

  for (int kt = 0; kt <= ktmax; kt++) {
    int kb = kt * 32;
    const __bf16* kbase0 = Kb + ((size_t)bh * SKV + (kb + lr)) * DH;
    const __bf16* kbase1 = kbase0 + (size_t)16 * DH;
    // batch all 8 K fragments, then run the WMMA chains (partial loadcnt waits)
    v16bf kf[8];
#pragma unroll
    for (int c = 0; c < 4; c++) {
      kf[c]     = load_frag(kbase0 + c * 32, half);
      kf[4 + c] = load_frag(kbase1 + c * 32, half);
    }
    v8f sc0 = vzero8(), sc1 = vzero8();
#pragma unroll
    for (int c = 0; c < 4; c++) sc0 = wmma_bf16(qf[c], kf[c], sc0);
#pragma unroll
    for (int c = 0; c < 4; c++) sc1 = wmma_bf16(qf[c], kf[4 + c], sc1);

    float p0[8], p1[8];
#pragma unroll
    for (int r = 0; r < 8; r++) {
      int i = m0 + r + 8 * half;
      int j0 = kb + lr, j1 = j0 + 16;
      bool ok0 = (j0 <= i) && !((i >= S_CTX) && (j0 >= S_CTX));
      bool ok1 = (j1 <= i) && !((i >= S_CTX) && (j1 >= S_CTX));
      p0[r] = sc0[r] * scale + (ok0 ? 0.f : -1.0e9f);
      p1[r] = sc1[r] * scale + (ok1 ? 0.f : -1.0e9f);
    }
#pragma unroll
    for (int r = 0; r < 8; r++) {
      float t = fmaxf(p0[r], p1[r]);
#pragma unroll
      for (int m = 1; m < 16; m <<= 1) t = fmaxf(t, __shfl_xor(t, m, 32));
      float mnew = fmaxf(mrun[r], t);
      float alpha = __expf(mrun[r] - mnew);
      mrun[r] = mnew;
      p0[r] = __expf(p0[r] - mnew);
      p1[r] = __expf(p1[r] - mnew);
      float rs = p0[r] + p1[r];
#pragma unroll
      for (int m = 1; m < 16; m <<= 1) rs += __shfl_xor(rs, m, 32);
      lrun[r] = lrun[r] * alpha + rs;
#pragma unroll
      for (int c = 0; c < 8; c++) oacc[c][r] *= alpha;
    }
    // C-layout (lane=N,vgpr=M) -> A-fragment via per-wave LDS tile
    __bf16* myP = &Pld[widx][0][0];
#pragma unroll
    for (int r = 0; r < 8; r++) {
      myP[(r + 8 * half) * 32 + lr]      = f2bf(p0[r]);
      myP[(r + 8 * half) * 32 + 16 + lr] = f2bf(p1[r]);
    }
    asm volatile("s_wait_dscnt 0" ::: "memory");
    v16bf pf = load_frag(myP + lr * 32, half);
    const __bf16* vbase = Vt + ((size_t)bh * DH + lr) * SKV + kb;
    // batch all 8 V fragments, then the WMMA chain
    v16bf vf[8];
#pragma unroll
    for (int c = 0; c < 8; c++) vf[c] = load_frag(vbase + (size_t)c * 16 * SKV, half);
#pragma unroll
    for (int c = 0; c < 8; c++) oacc[c] = wmma_bf16(pf, vf[c], oacc[c]);
  }
  // normalize + store to attn-out [b, s, h*128 + d] (bf16)
#pragma unroll
  for (int r = 0; r < 8; r++) {
    int m = m0 + r + 8 * half;
    if (m < S_TOT) {
      float inv = 1.0f / lrun[r];
      size_t rowo = ((size_t)b * S_TOT + m) * DMODEL + h * DH;
#pragma unroll
      for (int c = 0; c < 8; c++)
        AO[rowo + c * 16 + lr] = f2bf(oacc[c][r] * inv);
    }
  }
}

// ---------------- generic NT GEMM: C[m,n] = sum_k A[m,k]*B[n,k] + bias ----------------
__global__ void __launch_bounds__(128) gemm_nt(
    const __bf16* __restrict__ A, int lda,
    const __bf16* __restrict__ Bw, int ldb,
    const float* __restrict__ bias,
    float* __restrict__ Cf, __bf16* __restrict__ Cb, int ldc,
    int M, int N, int K, int relu) {
  int wid  = (blockIdx.x * blockDim.x + threadIdx.x) >> 5;
  int lane = threadIdx.x & 31;
  int half = lane >> 4, lr = lane & 15;
  int ntiles = (N + 63) >> 6;
  int mt = wid / ntiles, nt = wid % ntiles;
  if (mt * 16 >= M) return;
  int m0 = mt * 16, n0 = nt * 64;
  const __bf16* Arow = A + (size_t)(m0 + lr) * lda;
  const __bf16* Brow[4];
#pragma unroll
  for (int s = 0; s < 4; s++) Brow[s] = Bw + (size_t)(n0 + s * 16 + lr) * ldb;

  v8f acc[4];
#pragma unroll
  for (int s = 0; s < 4; s++) acc[s] = vzero8();

  // ping-pong double buffered K loop (nk is even for all K used here)
  int nk = K >> 5;
  v16bf aA = load_frag(Arow, half);
  v16bf bA[4], aB, bB[4];
#pragma unroll
  for (int s = 0; s < 4; s++)
    if (n0 + s * 16 < N) bA[s] = load_frag(Brow[s], half);
  for (int it = 0; it < nk; it += 2) {
    {
      int kc = (it + 1) * 32;
      aB = load_frag(Arow + kc, half);
#pragma unroll
      for (int s = 0; s < 4; s++)
        if (n0 + s * 16 < N) bB[s] = load_frag(Brow[s] + kc, half);
    }
#pragma unroll
    for (int s = 0; s < 4; s++)
      if (n0 + s * 16 < N) acc[s] = wmma_bf16(aA, bA[s], acc[s]);   // uniform guard
    if (it + 2 < nk) {
      int kc = (it + 2) * 32;
      aA = load_frag(Arow + kc, half);
#pragma unroll
      for (int s = 0; s < 4; s++)
        if (n0 + s * 16 < N) bA[s] = load_frag(Brow[s] + kc, half);
    }
#pragma unroll
    for (int s = 0; s < 4; s++)
      if (n0 + s * 16 < N) acc[s] = wmma_bf16(aB, bB[s], acc[s]);
  }

#pragma unroll
  for (int s = 0; s < 4; s++) {
    int nb = n0 + s * 16;
    if (nb >= N) continue;
    int n = nb + lr;
    float bv = bias ? bias[n] : 0.f;
#pragma unroll
    for (int r = 0; r < 8; r++) {
      int m = m0 + r + 8 * half;
      float v = acc[s][r] + bv;
      if (relu) v = fmaxf(v, 0.f);
      if (Cf) Cf[(size_t)m * ldc + n] = v;
      if (Cb) Cb[(size_t)m * ldc + n] = f2bf(v);
    }
  }
}

// ---------------- host orchestration ----------------
extern "C" void kernel_launch(void* const* d_in, const int* in_sizes, int n_in,
                              void* d_out, int out_size, void* d_ws, size_t ws_size,
                              hipStream_t stream) {
  (void)in_sizes; (void)n_in; (void)out_size; (void)ws_size;
  const int*   cat        = (const int*)d_in[0];
  const int*   tim        = (const int*)d_in[1];
  const int*   f0         = (const int*)d_in[2];
  const float* wcat       = (const float*)d_in[3];
  const float* pad        = (const float*)d_in[4];
  const float* attn_in_w  = (const float*)d_in[5];
  const float* attn_in_b  = (const float*)d_in[6];
  const float* attn_out_w = (const float*)d_in[7];
  const float* attn_out_b = (const float*)d_in[8];
  const float* ff1_w      = (const float*)d_in[9];
  const float* ff1_b      = (const float*)d_in[10];
  const float* ff2_w      = (const float*)d_in[11];
  const float* ff2_b      = (const float*)d_in[12];
  const float* ln1_g      = (const float*)d_in[13];
  const float* ln1_b      = (const float*)d_in[14];
  const float* ln2_g      = (const float*)d_in[15];
  const float* ln2_b      = (const float*)d_in[16];
  float* out = (float*)d_out;

  char* ws = (char*)d_ws;
  size_t off = 0;
  auto take = [&](size_t bytes) -> char* {
    char* p = ws + off;
    off = (off + bytes + 255) & ~(size_t)255;
    return p;
  };
  float*  XF  = (float*)take((size_t)ROWS * DMODEL * 4);
  __bf16* XB  = (__bf16*)take((size_t)ROWS * DMODEL * 2);
  size_t qkvElems = (size_t)NB * NH * SKV * DH;
  __bf16* Qb  = (__bf16*)take(qkvElems * 2 * 3);  // Q,K,Vt contiguous (one memset)
  __bf16* Kb  = Qb + qkvElems;
  __bf16* Vt  = Kb + qkvElems;
  __bf16* AO  = (__bf16*)take((size_t)ROWS * DMODEL * 2);
  __bf16* HM  = (__bf16*)take((size_t)ROWS * DFF * 2);
  float*  T32 = (float*)take((size_t)ROWS * DMODEL * 4);
  __bf16* WC  = (__bf16*)take((size_t)NUM_CAT * D_PART * 2);
  __bf16* WO  = (__bf16*)take((size_t)NUM_CONT * D_CONT * 2);
  __bf16* WIN = (__bf16*)take((size_t)2 * 3 * DMODEL * DMODEL * 2);
  __bf16* WOUT= (__bf16*)take((size_t)2 * DMODEL * DMODEL * 2);
  __bf16* WF1 = (__bf16*)take((size_t)2 * DFF * DMODEL * 2);
  __bf16* WF2 = (__bf16*)take((size_t)2 * DMODEL * DFF * 2);
  __bf16* HIN = (__bf16*)take((size_t)16 * DMODEL * 2);

  auto conv = [&](const float* src, __bf16* dst, size_t n) {
    conv_bf16<<<dim3((unsigned)((n + 255) / 256)), dim3(256), 0, stream>>>(src, dst, (int)n);
  };
  conv(wcat, WC, (size_t)NUM_CAT * D_PART);
  conv(attn_in_w,  WIN,  (size_t)2 * 3 * DMODEL * DMODEL);
  conv(attn_out_w, WOUT, (size_t)2 * DMODEL * DMODEL);
  conv(ff1_w, WF1, (size_t)2 * DFF * DMODEL);
  conv(ff2_w, WF2, (size_t)2 * DMODEL * DFF);
  gen_wcont<<<dim3((NUM_CONT * D_CONT + 255) / 256), dim3(256), 0, stream>>>(WO);
  build_x<<<dim3(ROWS), dim3(256), 0, stream>>>(cat, tim, f0, wcat, pad, XF, XB);
  hipMemsetAsync(Qb, 0, qkvElems * 2 * 3, stream);   // zero key pads (graph-safe)

  auto gemm = [&](const __bf16* A, int lda, const __bf16* Bm, int ldb,
                  const float* bias, float* Cf, __bf16* Cb, int ldc,
                  int M, int N, int K, int relu) {
    int waves = ((M + 15) / 16) * ((N + 63) / 64);
    gemm_nt<<<dim3((waves + 3) / 4), dim3(128), 0, stream>>>(
        A, lda, Bm, ldb, bias, Cf, Cb, ldc, M, N, K, relu);
  };

  for (int l = 0; l < 2; l++) {
    {
      int waves = (ROWS / 16) * ((3 * DMODEL) / 64);
      gemm_qkv<<<dim3((waves + 3) / 4), dim3(128), 0, stream>>>(
          XB, WIN + (size_t)l * 3 * DMODEL * DMODEL,
          attn_in_b + (size_t)l * 3 * DMODEL, Qb, Kb, Vt);
    }
    {
      int waves = NB * NH * ((S_TOT + 15) / 16);
      flash_attn<<<dim3((waves + 3) / 4), dim3(128), 0, stream>>>(Qb, Kb, Vt, AO);
    }
    gemm(AO, DMODEL, WOUT + (size_t)l * DMODEL * DMODEL, DMODEL,
         attn_out_b + (size_t)l * DMODEL, T32, nullptr, DMODEL,
         ROWS, DMODEL, DMODEL, 0);
    residual_ln<<<dim3(ROWS), dim3(256), 0, stream>>>(
        XF, T32, ln1_g + (size_t)l * DMODEL, ln1_b + (size_t)l * DMODEL, XF, XB);
    gemm(XB, DMODEL, WF1 + (size_t)l * DFF * DMODEL, DMODEL,
         ff1_b + (size_t)l * DFF, nullptr, HM, DFF, ROWS, DFF, DMODEL, 1);
    gemm(HM, DFF, WF2 + (size_t)l * DMODEL * DFF, DFF,
         ff2_b + (size_t)l * DMODEL, T32, nullptr, DMODEL, ROWS, DMODEL, DFF, 0);
    residual_ln<<<dim3(ROWS), dim3(256), 0, stream>>>(
        XF, T32, ln2_g + (size_t)l * DMODEL, ln2_b + (size_t)l * DMODEL, XF, XB);
  }

  head_gather<<<dim3((16 * DMODEL + 255) / 256), dim3(256), 0, stream>>>(XF, HIN);
  // pc @ w_cat.T  -> out[0 : 16*50000]
  gemm(HIN, DMODEL, WC, D_PART, nullptr, out, nullptr, NUM_CAT,
       16, NUM_CAT, D_PART, 0);
  // pt @ w_cont.T -> out[800000 : +16*4096]
  gemm(HIN + D_PART, DMODEL, WO, D_CONT, nullptr,
       out + (size_t)16 * NUM_CAT, nullptr, NUM_CONT, 16, NUM_CONT, D_CONT, 0);
  // pf @ w_cont.T -> out[865536 : +16*4096]
  gemm(HIN + D_PART + D_CONT, DMODEL, WO, D_CONT, nullptr,
       out + (size_t)16 * NUM_CAT + (size_t)16 * NUM_CONT, nullptr, NUM_CONT,
       16, NUM_CONT, D_CONT, 0);
}
